// SelfAttentionBlock_33878702030865
// MI455X (gfx1250) — matrix operand order repaired and checked by reading, not compiled, not run on hardware
//
#include <hip/hip_runtime.h>
#include <hip/hip_bf16.h>
#include <math.h>

// Problem constants from the reference
#define BB   2
#define NN   2048
#define DD   256
#define HH   8
#define DHH  32
#define HIDN 512
#define MM   (BB * NN)   // 4096 rows

typedef __attribute__((ext_vector_type(16))) __bf16 v16bf;
typedef __attribute__((ext_vector_type(8)))  float  v8f;

union BF16Frag {
  v16bf v;
  unsigned int   u32[8];
  unsigned short u16[16];
};

__device__ __forceinline__ v8f vzero8() {
  v8f c;
  #pragma unroll
  for (int i = 0; i < 8; ++i) c[i] = 0.0f;
  return c;
}

__device__ __forceinline__ unsigned short f2bf(float x) {
  unsigned int u = __float_as_uint(x);
  u += 0x7FFFu + ((u >> 16) & 1u);   // round-to-nearest-even
  return (unsigned short)(u >> 16);
}

// reductions across the 16 lanes of a half-wave (xor masks < 16 stay in-half)
__device__ __forceinline__ float hmax16(float v) {
  #pragma unroll
  for (int m = 1; m < 16; m <<= 1) v = fmaxf(v, __shfl_xor(v, m, 32));
  return v;
}
__device__ __forceinline__ float hsum16(float v) {
  #pragma unroll
  for (int m = 1; m < 16; m <<= 1) v += __shfl_xor(v, m, 32);
  return v;
}

// ---------------------------------------------------------------------------
// bf16 WMMA 16x64 output strip of  A[M,K] @ W[N,K]^T  (both row-major).
// One A fragment is reused across 4 adjacent 16-wide N tiles per k-step.
// All 5 fragments of a k-step are issued as distinct live values BEFORE the
// 4 WMMAs, and the k-loop is fully unrolled (K is a compile-time constant at
// every call site), so the scheduler can keep many b128 loads in flight and
// overlap them with the WMMA pipe instead of draining loadcnt per matmul.
// A fragment (contiguous k order, not AMD's table order): lane(l16)=row M, VGPR i holds K pair
//   ka = 16*hi + 2i                        -> contiguous 32-bit pair in A row.
// B fragment (same contiguous k order as A): lane(l16)=col N, VGPR r holds K pair
//   kb = 16*hi + 2r                        -> contiguous 32-bit pair in W row.
// ---------------------------------------------------------------------------
struct Acc4 { v8f c[4]; };

__device__ __forceinline__ Acc4 wmma_strip4(const unsigned short* __restrict__ A,
                                            const unsigned short* __restrict__ W,
                                            int m0, int n0, int K) {
  const int lane = threadIdx.x & 31;
  const int hi   = lane >> 4;
  const int l16  = lane & 15;
  Acc4 acc;
  #pragma unroll
  for (int t = 0; t < 4; ++t) acc.c[t] = vzero8();
  const unsigned int* arow = (const unsigned int*)(A + (size_t)(m0 + l16) * K);
  const unsigned int* wrow[4];
  #pragma unroll
  for (int t = 0; t < 4; ++t)
    wrow[t] = (const unsigned int*)(W + (size_t)(n0 + t * 16 + l16) * K);
  #pragma unroll
  for (int k0 = 0; k0 < K; k0 += 32) {
    BF16Frag a, b[4];
    #pragma unroll
    for (int i = 0; i < 8; ++i) {
      const int ka = 16 * hi + 2 * i;
      a.u32[i] = arow[(k0 + ka) >> 1];
    }
    #pragma unroll
    for (int t = 0; t < 4; ++t) {
      #pragma unroll
      for (int i = 0; i < 8; ++i) b[t].u32[i] = wrow[t][(k0 + 16 * hi + 2 * i) >> 1];
    }
    #pragma unroll
    for (int t = 0; t < 4; ++t) {
      acc.c[t] = __builtin_amdgcn_wmma_f32_16x16x32_bf16(false, a.v, false, b[t].v,
                                                         (short)0, acc.c[t], false, false);
    }
  }
  return acc;
}

// ---------------------------------------------------------------------------
// fp32 -> bf16 weight conversion
// ---------------------------------------------------------------------------
__global__ void __launch_bounds__(256) f2bf_kernel(const float* __restrict__ s,
                                                   unsigned short* __restrict__ d, int n) {
  const int i = blockIdx.x * 256 + threadIdx.x;
  if (i < n) d[i] = f2bf(s[i]);
}

// ---------------------------------------------------------------------------
// LayerNorm over D=256 (one row per 256-thread block), optional fp32 copy out
// and optional tau = x @ tau_w^T + tau_b head projection (8 dots).
// ---------------------------------------------------------------------------
__global__ void __launch_bounds__(256) ln_kernel(
    const float* __restrict__ src, const float* __restrict__ gamma,
    const float* __restrict__ beta, const float* __restrict__ tw,
    const float* __restrict__ tb, float* __restrict__ xf,
    unsigned short* __restrict__ xb, float* __restrict__ tau) {
  __shared__ float red[256];
  const int row = blockIdx.x;
  const int d   = threadIdx.x;
  const float v = src[(size_t)row * DD + d];
  red[d] = v;
  __syncthreads();
  #pragma unroll
  for (int s = 128; s > 0; s >>= 1) { if (d < s) red[d] += red[d + s]; __syncthreads(); }
  const float mean = red[0] * (1.0f / DD);
  __syncthreads();
  const float c = v - mean;
  red[d] = c * c;
  __syncthreads();
  #pragma unroll
  for (int s = 128; s > 0; s >>= 1) { if (d < s) red[d] += red[d + s]; __syncthreads(); }
  const float var = red[0] * (1.0f / DD);
  __syncthreads();
  const float y = c * rsqrtf(var + 1e-6f) * gamma[d] + beta[d];
  if (xf) xf[(size_t)row * DD + d] = y;
  xb[(size_t)row * DD + d] = f2bf(y);
  if (tau) {
    for (int h = 0; h < HH; ++h) {
      red[d] = y * tw[h * DD + d];
      __syncthreads();
      #pragma unroll
      for (int s = 128; s > 0; s >>= 1) { if (d < s) red[d] += red[d + s]; __syncthreads(); }
      if (d == 0) tau[(size_t)row * HH + h] = red[0] + tb[h];
      __syncthreads();
    }
  }
}

// ---------------------------------------------------------------------------
// QKV projection: [4096,256] @ [768,256]^T. Epilogue scatters q,k row-major
// bf16 and v transposed to [B,H,DH,N] so the PV matmul B-operand pairs are
// contiguous along the key dimension. One wave = 16x64 output strip.
// ---------------------------------------------------------------------------
__global__ void __launch_bounds__(256) gemm_qkv_kernel(
    const unsigned short* __restrict__ xb, const unsigned short* __restrict__ wbf,
    const float* __restrict__ bias, unsigned short* __restrict__ q,
    unsigned short* __restrict__ k, unsigned short* __restrict__ vT) {
  const int wave = blockIdx.x * 8 + (threadIdx.x >> 5);
  const int tm = wave / 12, tsn = wave % 12;   // 12 strips of 64 cover N=768
  const int lane = threadIdx.x & 31, hi = lane >> 4, l16 = lane & 15;
  Acc4 acc = wmma_strip4(xb, wbf, tm * 16, tsn * 64, DD);
  #pragma unroll
  for (int t = 0; t < 4; ++t) {
    #pragma unroll
    for (int i = 0; i < 8; ++i) {
      const int row = tm * 16 + i + 8 * hi;
      const int col = tsn * 64 + t * 16 + l16;
      const unsigned short bv = f2bf(acc.c[t][i] + bias[col]);
      if (col < DD) {
        q[(size_t)row * DD + col] = bv;
      } else if (col < 2 * DD) {
        k[(size_t)row * DD + (col - DD)] = bv;
      } else {
        const int j  = col - 2 * DD;
        const int dh = j & (DHH - 1), hh = j >> 5;
        const int b  = row >> 11, n = row & (NN - 1);
        vT[((size_t)(b * HH + hh) * DHH + dh) * NN + n] = bv;
      }
    }
  }
}

// ---------------------------------------------------------------------------
// Flash attention: 1 wave = 16 queries x 1 head, streaming 32-key chunks.
// scores tile C-layout: lane(l16)=key col, VGPR i = q row (i + 8*hi).
// Bias -sqrt(dx^2+dy^2)*tau[qrow] is fused per element; online softmax uses
// half-wave shfl reductions; P goes through a per-wave LDS slab to transpose
// from C-layout into A-layout for the P@V WMMAs.
// ---------------------------------------------------------------------------
__global__ void __launch_bounds__(128) attn_kernel(
    const unsigned short* __restrict__ qb, const unsigned short* __restrict__ kb,
    const unsigned short* __restrict__ vT, const float* __restrict__ tau,
    const float* __restrict__ pts, unsigned short* __restrict__ ctxb) {
  __shared__ unsigned short plds[4][16][32];
  const int wid  = threadIdx.x >> 5;
  const int lane = threadIdx.x & 31;
  const int hi   = lane >> 4;
  const int l16  = lane & 15;
  const int gw = blockIdx.x * 4 + wid;
  const int qt = gw & 127;          // q-tile within (b,h)
  const int bh = gw >> 7;           // 0..15
  const int h  = bh & (HH - 1);
  const int b  = bh >> 3;
  const int q0 = qt * 16;

  float qx[8], qy[8], tr[8], rm[8], rl[8];
  v8f ctx0 = vzero8(), ctx1 = vzero8();
  #pragma unroll
  for (int i = 0; i < 8; ++i) {
    const int row = q0 + i + 8 * hi;
    qx[i] = pts[((size_t)b * NN + row) * 3 + 0];
    qy[i] = pts[((size_t)b * NN + row) * 3 + 1];
    tr[i] = tau[((size_t)b * NN + row) * HH + h];
    rm[i] = -1e30f;
    rl[i] = 0.0f;
  }

  BF16Frag qa;   // Q tile A-fragment, loaded once (DH=32 == one WMMA K step)
  {
    const unsigned int* qrow =
        (const unsigned int*)(qb + (size_t)(b * NN + q0 + l16) * DD + h * DHH);
    #pragma unroll
    for (int i = 0; i < 8; ++i) {
      const int ka = 16 * hi + 2 * i;
      qa.u32[i] = qrow[ka >> 1];
    }
  }

  const float scale = 0.17677669529663689f;  // 1/sqrt(32)

  for (int kc = 0; kc < NN; kc += 32) {
    // ---- scores: Q[16x32] @ K^T[32x16] for two key sub-tiles ----
    v8f s0 = vzero8(), s1 = vzero8();
    {
      BF16Frag bk0, bk1;
      const unsigned int* kr0 =
          (const unsigned int*)(kb + (size_t)(b * NN + kc + l16) * DD + h * DHH);
      const unsigned int* kr1 =
          (const unsigned int*)(kb + (size_t)(b * NN + kc + 16 + l16) * DD + h * DHH);
      #pragma unroll
      for (int i = 0; i < 8; ++i) bk0.u32[i] = kr0[(16 * hi + 2 * i) >> 1];
      #pragma unroll
      for (int i = 0; i < 8; ++i) bk1.u32[i] = kr1[(16 * hi + 2 * i) >> 1];
      s0 = __builtin_amdgcn_wmma_f32_16x16x32_bf16(false, qa.v, false, bk0.v,
                                                   (short)0, s0, false, false);
      s1 = __builtin_amdgcn_wmma_f32_16x16x32_bf16(false, qa.v, false, bk1.v,
                                                   (short)0, s1, false, false);
    }

    // ---- fused distance bias + online softmax ----
    const int key0 = kc + l16, key1 = kc + 16 + l16;
    const float kx0 = pts[((size_t)b * NN + key0) * 3 + 0];
    const float ky0 = pts[((size_t)b * NN + key0) * 3 + 1];
    const float kx1 = pts[((size_t)b * NN + key1) * 3 + 0];
    const float ky1 = pts[((size_t)b * NN + key1) * 3 + 1];

    float sc0[8], sc1[8], cm[8];
    #pragma unroll
    for (int i = 0; i < 8; ++i) {
      const float dx0 = qx[i] - kx0, dy0 = qy[i] - ky0;
      const float dx1 = qx[i] - kx1, dy1 = qy[i] - ky1;
      sc0[i] = s0[i] * scale - sqrtf(dx0 * dx0 + dy0 * dy0) * tr[i];
      sc1[i] = s1[i] * scale - sqrtf(dx1 * dx1 + dy1 * dy1) * tr[i];
      cm[i]  = fmaxf(sc0[i], sc1[i]);
    }
    #pragma unroll
    for (int i = 0; i < 8; ++i) cm[i] = hmax16(cm[i]);

    float ps[8];
    #pragma unroll
    for (int i = 0; i < 8; ++i) {
      const float mn   = fmaxf(rm[i], cm[i]);
      const float corr = __expf(rm[i] - mn);
      sc0[i] = __expf(sc0[i] - mn);
      sc1[i] = __expf(sc1[i] - mn);
      ps[i]  = sc0[i] + sc1[i];
      rm[i]  = mn;
      rl[i] *= corr;
      ctx0[i] *= corr;
      ctx1[i] *= corr;
    }
    #pragma unroll
    for (int i = 0; i < 8; ++i) rl[i] += hsum16(ps[i]);

    // ---- transpose P (C-layout -> A-layout) through per-wave LDS slab ----
    #pragma unroll
    for (int i = 0; i < 8; ++i) {
      plds[wid][i + 8 * hi][l16]      = f2bf(sc0[i]);
      plds[wid][i + 8 * hi][16 + l16] = f2bf(sc1[i]);
    }
    __asm__ volatile("" ::: "memory");  // keep LDS store->load order (same wave)

    BF16Frag pa, bv0, bv1;
    const unsigned int* prow = (const unsigned int*)(&plds[wid][l16][0]);
    #pragma unroll
    for (int i = 0; i < 8; ++i) {
      const int ka = 16 * hi + 2 * i;
      pa.u32[i] = prow[ka >> 1];
    }
    // ---- ctx += P[16x32] @ V[32keys x 32dh], two dh sub-tiles ----
    const unsigned int* vr0 =
        (const unsigned int*)(vT + ((size_t)(b * HH + h) * DHH + l16) * NN + kc);
    const unsigned int* vr1 =
        (const unsigned int*)(vT + ((size_t)(b * HH + h) * DHH + 16 + l16) * NN + kc);
    #pragma unroll
    for (int i = 0; i < 8; ++i) bv0.u32[i] = vr0[(16 * hi + 2 * i) >> 1];
    #pragma unroll
    for (int i = 0; i < 8; ++i) bv1.u32[i] = vr1[(16 * hi + 2 * i) >> 1];
    ctx0 = __builtin_amdgcn_wmma_f32_16x16x32_bf16(false, pa.v, false, bv0.v,
                                                   (short)0, ctx0, false, false);
    ctx1 = __builtin_amdgcn_wmma_f32_16x16x32_bf16(false, pa.v, false, bv1.v,
                                                   (short)0, ctx1, false, false);
    __asm__ volatile("" ::: "memory");  // don't let next chunk's P stores hoist
  }

  #pragma unroll
  for (int i = 0; i < 8; ++i) {
    const float inv  = 1.0f / rl[i];
    const size_t row = (size_t)b * NN + q0 + i + 8 * hi;
    ctxb[row * DD + h * DHH + l16]      = f2bf(ctx0[i] * inv);
    ctxb[row * DD + h * DHH + 16 + l16] = f2bf(ctx1[i] * inv);
  }
}

// ---------------------------------------------------------------------------
// Output projection + both residuals: x2 = LN(x) + ctx@out_w^T + out_b + feat
// ---------------------------------------------------------------------------
__global__ void __launch_bounds__(256) gemm_out_kernel(
    const unsigned short* __restrict__ ctxb, const unsigned short* __restrict__ wbf,
    const float* __restrict__ bias, const float* __restrict__ xf,
    const float* __restrict__ feat, float* __restrict__ x2) {
  const int wave = blockIdx.x * 8 + (threadIdx.x >> 5);
  const int tm = wave / 4, tsn = wave % 4;     // 4 strips of 64 cover N=256
  const int lane = threadIdx.x & 31, hi = lane >> 4, l16 = lane & 15;
  Acc4 acc = wmma_strip4(ctxb, wbf, tm * 16, tsn * 64, DD);
  #pragma unroll
  for (int t = 0; t < 4; ++t) {
    #pragma unroll
    for (int i = 0; i < 8; ++i) {
      const int row = tm * 16 + i + 8 * hi;
      const int col = tsn * 64 + t * 16 + l16;
      const size_t idx = (size_t)row * DD + col;
      x2[idx] = acc.c[t][i] + bias[col] + xf[idx] + feat[idx];
    }
  }
}

// ---------------------------------------------------------------------------
// FFN fc1 with exact-erf GELU epilogue
// ---------------------------------------------------------------------------
__global__ void __launch_bounds__(256) gemm_fc1_kernel(
    const unsigned short* __restrict__ yb, const unsigned short* __restrict__ wbf,
    const float* __restrict__ bias, unsigned short* __restrict__ hb) {
  const int wave = blockIdx.x * 8 + (threadIdx.x >> 5);
  const int tm = wave / 8, tsn = wave % 8;     // 8 strips of 64 cover N=512
  const int lane = threadIdx.x & 31, hi = lane >> 4, l16 = lane & 15;
  Acc4 acc = wmma_strip4(yb, wbf, tm * 16, tsn * 64, DD);
  #pragma unroll
  for (int t = 0; t < 4; ++t) {
    #pragma unroll
    for (int i = 0; i < 8; ++i) {
      const int row = tm * 16 + i + 8 * hi;
      const int col = tsn * 64 + t * 16 + l16;
      const float v = acc.c[t][i] + bias[col];
      const float g = 0.5f * v * (1.0f + erff(v * 0.70710678118654752f));
      hb[(size_t)row * HIDN + col] = f2bf(g);
    }
  }
}

// ---------------------------------------------------------------------------
// FFN fc2 -> final fp32 output
// ---------------------------------------------------------------------------
__global__ void __launch_bounds__(256) gemm_fc2_kernel(
    const unsigned short* __restrict__ hb, const unsigned short* __restrict__ wbf,
    const float* __restrict__ bias, float* __restrict__ out) {
  const int wave = blockIdx.x * 8 + (threadIdx.x >> 5);
  const int tm = wave / 4, tsn = wave % 4;     // 4 strips of 64 cover N=256
  const int lane = threadIdx.x & 31, hi = lane >> 4, l16 = lane & 15;
  Acc4 acc = wmma_strip4(hb, wbf, tm * 16, tsn * 64, HIDN);
  #pragma unroll
  for (int t = 0; t < 4; ++t) {
    #pragma unroll
    for (int i = 0; i < 8; ++i) {
      const int row = tm * 16 + i + 8 * hi;
      const int col = tsn * 64 + t * 16 + l16;
      out[(size_t)row * DD + col] = acc.c[t][i] + bias[col];
    }
  }
}

// ---------------------------------------------------------------------------
extern "C" void kernel_launch(void* const* d_in, const int* in_sizes, int n_in,
                              void* d_out, int out_size, void* d_ws, size_t ws_size,
                              hipStream_t stream) {
  (void)in_sizes; (void)n_in; (void)out_size; (void)ws_size;
  const float* feat  = (const float*)d_in[0];
  const float* pts   = (const float*)d_in[1];
  const float* qn_g  = (const float*)d_in[2];
  const float* qn_b  = (const float*)d_in[3];
  const float* tau_w = (const float*)d_in[4];
  const float* tau_b = (const float*)d_in[5];
  const float* in_w  = (const float*)d_in[6];
  const float* in_b  = (const float*)d_in[7];
  const float* out_w = (const float*)d_in[8];
  const float* out_b = (const float*)d_in[9];
  const float* fn_g  = (const float*)d_in[10];
  const float* fn_b  = (const float*)d_in[11];
  const float* fc1_w = (const float*)d_in[12];
  const float* fc1_b = (const float*)d_in[13];
  const float* fc2_w = (const float*)d_in[14];
  const float* fc2_b = (const float*)d_in[15];
  float* out = (float*)d_out;

  // Workspace carve-out (~26.5 MB total)
  char* w = (char*)d_ws;
  size_t off = 0;
  auto carve = [&](size_t bytes) {
    void* p = w + off;
    off += (bytes + 255) & ~(size_t)255;
    return p;
  };
  float*          xf   = (float*)carve((size_t)MM * DD * 4);     // LN(feat) fp32
  unsigned short* xb   = (unsigned short*)carve((size_t)MM * DD * 2);
  float*          tau  = (float*)carve((size_t)MM * HH * 4);
  unsigned short* qb   = (unsigned short*)carve((size_t)MM * DD * 2);
  unsigned short* kbf  = (unsigned short*)carve((size_t)MM * DD * 2);
  unsigned short* vT   = (unsigned short*)carve((size_t)MM * DD * 2);
  unsigned short* ctx  = (unsigned short*)carve((size_t)MM * DD * 2);
  float*          x2   = (float*)carve((size_t)MM * DD * 4);
  unsigned short* yb   = (unsigned short*)carve((size_t)MM * DD * 2);
  unsigned short* hb   = (unsigned short*)carve((size_t)MM * HIDN * 2);
  unsigned short* winb = (unsigned short*)carve((size_t)3 * DD * DD * 2);
  unsigned short* wob  = (unsigned short*)carve((size_t)DD * DD * 2);
  unsigned short* w1b  = (unsigned short*)carve((size_t)HIDN * DD * 2);
  unsigned short* w2b  = (unsigned short*)carve((size_t)DD * HIDN * 2);

  // 1) weights -> bf16
  int n;
  n = 3 * DD * DD; f2bf_kernel<<<(n + 255) / 256, 256, 0, stream>>>(in_w, winb, n);
  n = DD * DD;     f2bf_kernel<<<(n + 255) / 256, 256, 0, stream>>>(out_w, wob, n);
  n = HIDN * DD;   f2bf_kernel<<<(n + 255) / 256, 256, 0, stream>>>(fc1_w, w1b, n);
  n = DD * HIDN;   f2bf_kernel<<<(n + 255) / 256, 256, 0, stream>>>(fc2_w, w2b, n);

  // 2) LN + tau head projection
  ln_kernel<<<MM, 256, 0, stream>>>(feat, qn_g, qn_b, tau_w, tau_b, xf, xb, tau);

  // 3) QKV projection (WMMA, 16x64 strips): 256 m-tiles * 12 strips / 8 waves
  gemm_qkv_kernel<<<(MM / 16) * 12 / 8, 256, 0, stream>>>(xb, winb, in_b, qb, kbf, vT);

  // 4) flash attention with fused distance bias (WMMA)
  attn_kernel<<<(BB * HH * (NN / 16)) / 4, 128, 0, stream>>>(qb, kbf, vT, tau, pts, ctx);

  // 5) out-proj + both residuals (WMMA): 256 * 4 / 8
  gemm_out_kernel<<<(MM / 16) * 4 / 8, 256, 0, stream>>>(ctx, wob, out_b, xf, feat, x2);

  // 6) second LN
  ln_kernel<<<MM, 256, 0, stream>>>(x2, fn_g, fn_b, nullptr, nullptr, nullptr, yb, nullptr);

  // 7) FFN (WMMA): fc1 256*8/8, fc2 256*4/8
  gemm_fc1_kernel<<<(MM / 16) * 8 / 8, 256, 0, stream>>>(yb, w1b, fc1_b, hb);
  gemm_fc2_kernel<<<(MM / 16) * 4 / 8, 256, 0, stream>>>(hb, w2b, fc2_b, out);
}
